// ChebyshevSelfAttention_17712445129301
// MI455X (gfx1250) — matrix operand order repaired and checked by reading, not compiled, not run on hardware
//
#include <hip/hip_runtime.h>
#include <hip/hip_bf16.h>

#define HIDDEN 768
#define NHEADS 12
#define HDIM   64
#define ORDER  5

typedef __bf16 bf16;
typedef bf16  v16bf __attribute__((ext_vector_type(16)));
typedef float v8f   __attribute__((ext_vector_type(8)));
typedef int   aint4 __attribute__((vector_size(16)));   // matches builtin param

#define AS1 __attribute__((address_space(1)))
#define AS3 __attribute__((address_space(3)))

#if defined(__HIP_DEVICE_COMPILE__) && __has_builtin(__builtin_amdgcn_global_load_async_to_lds_b128)
#define HAVE_ASYNC_LDS 1
#else
#define HAVE_ASYNC_LDS 0
#endif

union ABFrag { v16bf v; unsigned int u[8]; };

// ---------------------------------------------------------------------------
// WMMA helpers (CDNA5 wave32 layouts, CDNA5 ISA §7.12.2)
// ---------------------------------------------------------------------------

__device__ __forceinline__ v8f wmma_bf16(v16bf a, v16bf b, v8f c) {
  return __builtin_amdgcn_wmma_f32_16x16x32_bf16(
      /*neg_a=*/false, a, /*neg_b=*/false, b,
      /*c_mod=*/(short)0, c, /*reuse_a=*/false, /*reuse_b=*/false);
}

// A fragment (16x32 bf16).  base points at tile element (row 0, k 0),
// row stride = ld elements.  Lane l: row = l&15; lanes 0-15 hold K 0-15,
// lanes 16-31 hold K 16-31; each VGPR = 2 consecutive K -> dword loads.
__device__ __forceinline__ v16bf load_frag_a(const bf16* base, int ld) {
  const int l  = threadIdx.x & 31;
  const int m  = l & 15;
  const int k2 = (l < 16) ? 0 : 8;          // dword offset of K base (0 or 16 elems)
  const unsigned int* p = (const unsigned int*)(base + m * ld);
  ABFrag f;
#pragma unroll
  for (int v = 0; v < 4; ++v) f.u[v] = p[k2 + v];
#pragma unroll
  for (int v = 0; v < 4; ++v) f.u[4 + v] = p[k2 + 4 + v];
  return f.v;
}

// B fragment (32x16 bf16) where B[k][n] = base[n*ld + k] (source rows are the
// N dimension, K contiguous).  Lane l: n = l&15; lanes 0-15 hold K 0-15,
// lanes 16-31 hold K 16-31; consecutive K pairs -> dword loads.
__device__ __forceinline__ v16bf load_frag_b(const bf16* base, int ld) {
  const int l  = threadIdx.x & 31;
  const int n  = l & 15;
  const int k2 = (l < 16) ? 0 : 8;          // dword offset of K base (0 or 16 elems)
  const unsigned int* p = (const unsigned int*)(base + n * ld);
  ABFrag f;
#pragma unroll
  for (int v = 0; v < 8; ++v) f.u[v] = p[k2 + v];
  return f.v;
}

__device__ __forceinline__ float rmax16(float x) {
  x = fmaxf(x, __shfl_xor(x, 1, 32));
  x = fmaxf(x, __shfl_xor(x, 2, 32));
  x = fmaxf(x, __shfl_xor(x, 4, 32));
  x = fmaxf(x, __shfl_xor(x, 8, 32));
  return x;
}
__device__ __forceinline__ float rsum16(float x) {
  x += __shfl_xor(x, 1, 32);
  x += __shfl_xor(x, 2, 32);
  x += __shfl_xor(x, 4, 32);
  x += __shfl_xor(x, 8, 32);
  return x;
}

__device__ __forceinline__ void wait_async0() {
#if HAVE_ASYNC_LDS
#if __has_builtin(__builtin_amdgcn_s_wait_asynccnt)
  __builtin_amdgcn_s_wait_asynccnt(0);
#else
  __asm__ volatile("s_wait_asynccnt 0x0" ::: "memory");
#endif
#endif
}

// ---------------------------------------------------------------------------
// Kernel 1: fused QKV projection.  out[z][b,h,s,d] (bf16) = X @ W[z]^T + bias.
// Block tile 128(M) x 64(N), 8 waves as 4x2 grid of 32x32 sub-tiles, K-step 32.
// grid = (M/128, NHEADS, 3)  -> each (y) tile is exactly one head.
// ---------------------------------------------------------------------------
__global__ __launch_bounds__(256)
void qkv_proj_kernel(const float* __restrict__ X,
                     const float* __restrict__ Wq, const float* __restrict__ bq,
                     const float* __restrict__ Wk, const float* __restrict__ bk,
                     const float* __restrict__ Wv, const float* __restrict__ bv,
                     bf16* __restrict__ qkv, int S, int M) {
  __shared__ __align__(16) bf16 Xs[128 * 32];
  __shared__ __align__(16) bf16 Ws[64 * 32];

  const int tid  = threadIdx.x;
  const int wave = tid >> 5;
  const int wr   = wave & 3;           // 32-row sub-tile
  const int wc   = wave >> 2;          // 32-col sub-tile
  const int m0   = blockIdx.x * 128;
  const int h    = blockIdx.y;         // head == 64-wide N tile
  const int z    = blockIdx.z;         // 0:Q 1:K 2:V
  const int n0   = h * HDIM;

  const float* W    = (z == 0) ? Wq : (z == 1) ? Wk : Wv;
  const float* bvec = (z == 0) ? bq : (z == 1) ? bk : bv;

  v8f acc[2][2] = {};

  for (int kk = 0; kk < HIDDEN; kk += 32) {
    __syncthreads();
    // X tile 128x32 fp32 -> bf16 LDS (16 contiguous elems / thread)
    {
      const int row = tid >> 1;
      const int col = (tid & 1) * 16;
      const float* src = X + (size_t)(m0 + row) * HIDDEN + kk + col;
#pragma unroll
      for (int e = 0; e < 16; ++e) Xs[row * 32 + col + e] = (bf16)src[e];
    }
    // W tile 64x32 (rows = output n, cols = k) fp32 -> bf16 LDS
    {
      const int row = tid >> 2;
      const int col = (tid & 3) * 8;
      const float* src = W + (size_t)(n0 + row) * HIDDEN + kk + col;
#pragma unroll
      for (int e = 0; e < 8; ++e) Ws[row * 32 + col + e] = (bf16)src[e];
    }
    __syncthreads();

    v16bf a0 = load_frag_a(Xs + (wr * 32 +  0) * 32, 32);
    v16bf a1 = load_frag_a(Xs + (wr * 32 + 16) * 32, 32);
    v16bf b0 = load_frag_b(Ws + (wc * 32 +  0) * 32, 32);  // B[k][n]=W[n][k]
    v16bf b1 = load_frag_b(Ws + (wc * 32 + 16) * 32, 32);
    acc[0][0] = wmma_bf16(a0, b0, acc[0][0]);
    acc[0][1] = wmma_bf16(a0, b1, acc[0][1]);
    acc[1][0] = wmma_bf16(a1, b0, acc[1][0]);
    acc[1][1] = wmma_bf16(a1, b1, acc[1][1]);
  }

  // Epilogue: +bias, bf16, scatter to [z][b,h,s,d].
  const int l    = tid & 31;
  const int nl   = l & 15;
  const int hi8  = (l < 16) ? 0 : 8;
  const int bIdx = m0 / S;             // block never straddles a batch (S%128==0)
  const int sB   = m0 % S;
  bf16* outz = qkv + (size_t)z * M * HIDDEN
             + (((size_t)bIdx * NHEADS + h) * S) * HDIM;
#pragma unroll
  for (int jt = 0; jt < 2; ++jt) {
    const int d = wc * 32 + jt * 16 + nl;
    const float bb = bvec[n0 + d];
#pragma unroll
    for (int it = 0; it < 2; ++it) {
#pragma unroll
      for (int v = 0; v < 8; ++v) {
        const int s = sB + wr * 32 + it * 16 + hi8 + v;
        outz[(size_t)s * HDIM + d] = (bf16)(acc[it][jt][v] + bb);
      }
    }
  }
}

// ---------------------------------------------------------------------------
// Kernel 2: flash attention with Chebyshev relative-position bias.
// grid = (S/128, B*NHEADS); 8 waves x 16-row query strips; 64-key chunks.
// Software-pipelined: K chunk c+1 prefetched global->LDS asynchronously
// (GLOBAL_LOAD_ASYNC_TO_LDS_B128, ASYNCcnt) while chunk c is computed; V
// (needs a 16-bit transpose) is prefetched into registers before compute and
// drained to LDS after, so its load-wait also lands behind the WMMAs.
// ---------------------------------------------------------------------------
__global__ __launch_bounds__(256)
void cheb_attn_kernel(const bf16* __restrict__ qkv,
                      const float* __restrict__ alphas,
                      float* __restrict__ out, int B, int S) {
  __shared__ __align__(16) bf16 KsL[2][64 * 64];     // K chunk, [j][d]
  __shared__ __align__(16) bf16 VtL[2][64 * 64];     // V chunk transposed, [d][j]
  __shared__ __align__(16) bf16 PsL[8 * 16 * 64];    // per-wave P strips
  __shared__ float biasTab[2][192];                  // bias(h, j-i) window

  const int tid  = threadIdx.x;
  const int wave = tid >> 5;
  const int l    = tid & 31;
  const int nl   = l & 15;
  const int hi8  = (l < 16) ? 0 : 8;

  const int q0 = blockIdx.x * 128;
  const int bh = blockIdx.y;
  const int b  = bh / NHEADS;
  const int h  = bh % NHEADS;

  const size_t matSz = (size_t)B * NHEADS * S * HDIM;
  const bf16* Qg = qkv              + ((size_t)bh * S) * HDIM;
  const bf16* Kg = qkv +     matSz  + ((size_t)bh * S) * HDIM;
  const bf16* Vg = qkv + 2 * matSz  + ((size_t)bh * S) * HDIM;

  float al[ORDER + 1];
#pragma unroll
  for (int k = 0; k <= ORDER; ++k) al[k] = alphas[h * (ORDER + 1) + k];

  // Per-thread staging assignments.
  const int kByte = tid * 32;                 // K: 32 contiguous bytes / thread
  const int vRow  = tid >> 2;                 // V: row j, 16 cols from d0
  const int vCol  = (tid & 3) * 16;
  const float invs = 1.0f / (float)(S - 1);

  // Q A-fragments for this wave's strip live in registers for the whole pass.
  const bf16* qbase = Qg + (size_t)(q0 + wave * 16) * HDIM;
  const v16bf aq0 = load_frag_a(qbase,      HDIM);
  const v16bf aq1 = load_frag_a(qbase + 32, HDIM);

  v8f   o[4] = {};
  float run_m[8], run_l[8];
#pragma unroll
  for (int v = 0; v < 8; ++v) { run_m[v] = -INFINITY; run_l[v] = 0.0f; }

  bf16* Psw = PsL + wave * (16 * 64);
  const int nchunk = S / 64;

  // -------- prologue: stage chunk 0 into buffer 0 --------
  {
    const uint4* gk = (const uint4*)(Kg);
    ((uint4*)&KsL[0][0])[2 * tid + 0] = gk[2 * tid + 0];
    ((uint4*)&KsL[0][0])[2 * tid + 1] = gk[2 * tid + 1];
    const bf16* vsrc = Vg + (size_t)vRow * HDIM + vCol;
#pragma unroll
    for (int e = 0; e < 16; ++e) VtL[0][(vCol + e) * 64 + vRow] = vsrc[e];
    if (tid < 191) {                   // idx = (j - i) - (j0 - q0) + 127
      const int delta = tid - 127 + (0 - q0);
      float x = fminf(1.0f, fmaxf(-1.0f, (float)delta * invs));
      const float tx = 2.0f * x;
      float t0 = 1.0f, t1 = x;
      float r = al[0] + al[1] * x;
#pragma unroll
      for (int k = 2; k <= ORDER; ++k) {
        const float t2 = tx * t1 - t0;
        r += al[k] * t2;
        t0 = t1; t1 = t2;
      }
      biasTab[0][tid] = r;
    }
  }
  __syncthreads();

  for (int c = 0; c < nchunk; ++c) {
    const int cur = c & 1, nxt = cur ^ 1;
    const bool pf = (c + 1) < nchunk;

    // --- issue prefetch of chunk c+1 (completes during compute) ---
    bf16 vreg[16];
#if !HAVE_ASYNC_LDS
    uint4 kreg0, kreg1;
#endif
    if (pf) {
      const int j1 = (c + 1) * 64;
#if HAVE_ASYNC_LDS
      {
        const char* gk = (const char*)(Kg + (size_t)j1 * HDIM) + kByte;
        char* lk = (char*)&KsL[nxt][0] + kByte;
        __builtin_amdgcn_global_load_async_to_lds_b128(
            (AS1 aint4*)gk, (AS3 aint4*)lk, 0, 0);
        __builtin_amdgcn_global_load_async_to_lds_b128(
            (AS1 aint4*)gk, (AS3 aint4*)lk, 16, 0);
      }
#else
      const uint4* gk = (const uint4*)(Kg + (size_t)j1 * HDIM);
      kreg0 = gk[2 * tid + 0];
      kreg1 = gk[2 * tid + 1];
#endif
      const bf16* vsrc = Vg + (size_t)(j1 + vRow) * HDIM + vCol;
#pragma unroll
      for (int e = 0; e < 16; ++e) vreg[e] = vsrc[e];
    }

    // --- scores: 16x64 strip = Q(16x64) . K(64x64)^T, scaled + biased ---
    const bf16*  Ks = &KsL[cur][0];
    const bf16*  Vt = &VtL[cur][0];
    const float* bT = &biasTab[cur][0];
    float sc[4][8];
#pragma unroll
    for (int nt = 0; nt < 4; ++nt) {
      v8f s = {};
      s = wmma_bf16(aq0, load_frag_b(Ks + (nt * 16) * 64,      64), s);
      s = wmma_bf16(aq1, load_frag_b(Ks + (nt * 16) * 64 + 32, 64), s);
#pragma unroll
      for (int v = 0; v < 8; ++v) {
        const int idx = (nt * 16 + nl) - (wave * 16 + hi8 + v) + 127;
        sc[nt][v] = s[v] * 0.125f + bT[idx];
      }
    }

    // --- online softmax (row stats live in the 16-lane half-wave) ---
#pragma unroll
    for (int v = 0; v < 8; ++v) {
      float mv = fmaxf(fmaxf(sc[0][v], sc[1][v]), fmaxf(sc[2][v], sc[3][v]));
      mv = rmax16(mv);
      const float nm   = fmaxf(run_m[v], mv);
      const float corr = __expf(run_m[v] - nm);
      run_m[v] = nm;
      float lsum = 0.0f;
#pragma unroll
      for (int nt = 0; nt < 4; ++nt) {
        const float p = __expf(sc[nt][v] - nm);
        sc[nt][v] = p;
        lsum += p;
      }
      lsum = rsum16(lsum);
      run_l[v] = run_l[v] * corr + lsum;
#pragma unroll
      for (int nt = 0; nt < 4; ++nt) o[nt][v] *= corr;
    }

    // --- relayout P (C layout -> A layout) via per-wave LDS scratch ---
#pragma unroll
    for (int nt = 0; nt < 4; ++nt)
#pragma unroll
      for (int v = 0; v < 8; ++v)
        Psw[(hi8 + v) * 64 + nt * 16 + nl] = (bf16)sc[nt][v];

    __asm__ volatile("s_wait_dscnt 0" ::: "memory");  // LDS RAW fence (same wave)

    const v16bf ap0 = load_frag_a(Psw,      64);
    const v16bf ap1 = load_frag_a(Psw + 32, 64);
#pragma unroll
    for (int nt = 0; nt < 4; ++nt) {   // O += P(16x64) . V(64x64); B[j][d]=Vt[d][j]
      o[nt] = wmma_bf16(ap0, load_frag_b(Vt + (nt * 16) * 64,      64), o[nt]);
      o[nt] = wmma_bf16(ap1, load_frag_b(Vt + (nt * 16) * 64 + 32, 64), o[nt]);
    }

    // --- drain prefetch into buffer nxt (load-waits land after the WMMAs) ---
    if (pf) {
#if !HAVE_ASYNC_LDS
      ((uint4*)&KsL[nxt][0])[2 * tid + 0] = kreg0;
      ((uint4*)&KsL[nxt][0])[2 * tid + 1] = kreg1;
#endif
#pragma unroll
      for (int e = 0; e < 16; ++e) VtL[nxt][(vCol + e) * 64 + vRow] = vreg[e];
      if (tid < 191) {
        const int delta = tid - 127 + ((c + 1) * 64 - q0);
        float x = fminf(1.0f, fmaxf(-1.0f, (float)delta * invs));
        const float tx = 2.0f * x;
        float t0 = 1.0f, t1 = x;
        float r = al[0] + al[1] * x;
#pragma unroll
        for (int k = 2; k <= ORDER; ++k) {
          const float t2 = tx * t1 - t0;
          r += al[k] * t2;
          t0 = t1; t1 = t2;
        }
        biasTab[nxt][tid] = r;
      }
    }
    wait_async0();
    __syncthreads();
  }

  // --- normalize and store fp32 ctx at [b, s, h, d] ---
#pragma unroll
  for (int nt = 0; nt < 4; ++nt) {
#pragma unroll
    for (int v = 0; v < 8; ++v) {
      const int s = q0 + wave * 16 + hi8 + v;
      const int d = nt * 16 + nl;
      out[((size_t)b * S + s) * HIDDEN + h * HDIM + d] = o[nt][v] / run_l[v];
    }
  }
}

// ---------------------------------------------------------------------------
extern "C" void kernel_launch(void* const* d_in, const int* in_sizes, int n_in,
                              void* d_out, int out_size, void* d_ws, size_t ws_size,
                              hipStream_t stream) {
  const float* X  = (const float*)d_in[0];
  const float* Wq = (const float*)d_in[1];
  const float* bq = (const float*)d_in[2];
  const float* Wk = (const float*)d_in[3];
  const float* bk = (const float*)d_in[4];
  const float* Wv = (const float*)d_in[5];
  const float* bv = (const float*)d_in[6];
  const float* al = (const float*)d_in[7];
  float* out = (float*)d_out;

  const int M = in_sizes[0] / HIDDEN;  // B*S = 8192
  const int B = 4;
  const int S = M / B;                 // 2048

  // Workspace: Q,K,V in bf16, [3][B,H,S,D] = 3*M*768*2 B ~= 36 MB.
  bf16* qkv = (bf16*)d_ws;

  dim3 g1(M / 128, NHEADS, 3);
  qkv_proj_kernel<<<g1, 256, 0, stream>>>(X, Wq, bq, Wk, bk, Wv, bv, qkv, S, M);

  dim3 g2(S / 128, B * NHEADS);
  cheb_attn_kernel<<<g2, 256, 0, stream>>>(qkv, al, out, B, S);
}
